// GCN_28123445854600
// MI455X (gfx1250) — matrix operand order repaired and checked by reading, compiled, not run on hardware
//
#include <hip/hip_runtime.h>
#include <hip/hip_bf16.h>

// ---------------------------------------------------------------------------
// GCN (4x GCNConv+BN+ReLU, JK-sum heads, log_softmax) for gfx1250 / MI455X.
// GEMMs use v_wmma_f32_16x16x32_bf16 with bf16x3 error-compensated split,
// issued as 3 independent accumulator chains to avoid WMMA->WMMA RAW stalls.
// ---------------------------------------------------------------------------

typedef __attribute__((ext_vector_type(16))) __bf16 v16bf;
typedef __attribute__((ext_vector_type(8)))  float  v8f;

#define NLAYERS 4
#define FDIM    128
#define CDIM    40
#define CPAD    48   // padded head width (3 x 16 WMMA tiles)

// ---------------------------------------------------------------------------
// Degree / normalization
// ---------------------------------------------------------------------------
__global__ void deg_init_k(float* deg, int n) {
    int i = blockIdx.x * blockDim.x + threadIdx.x;
    if (i < n) deg[i] = 2.0f;  // improved=True self-loop weight 2
}

__global__ void deg_acc_k(const int* __restrict__ dst, float* deg, int e) {
    int i = blockIdx.x * blockDim.x + threadIdx.x;
    if (i < e) atomicAdd(&deg[dst[i]], 1.0f);
}

__global__ void deg_inv_k(float* deg, int n) {
    int i = blockIdx.x * blockDim.x + threadIdx.x;
    if (i < n) deg[i] = rsqrtf(deg[i]);  // deg >= 2 always
}

// ---------------------------------------------------------------------------
// Pack weights [K x ncols] (f32) into WMMA B-fragment order, bf16 hi/lo.
// Fragment element i (0..15) of lane L of tile (kt,nt):
//   k = kt*32 + (L>>4)*8 + (i&7) + (i>>3)*16 ,  n = nt*16 + (L&15)
// ---------------------------------------------------------------------------
__global__ void pack_w_k(const float* __restrict__ W, __bf16* __restrict__ hi,
                         __bf16* __restrict__ lo, int ncols, int NT,
                         int nlayers, int wstride) {
    int idx = blockIdx.x * blockDim.x + threadIdx.x;
    int per_layer = 4 * NT * 32 * 16;  // K=128 -> 4 kt tiles
    if (idx >= nlayers * per_layer) return;
    int l = idx / per_layer;
    int r = idx % per_layer;
    int i    = r & 15;
    int lane = (r >> 4) & 31;
    int nt   = (r >> 9) % NT;
    int kt   = (r >> 9) / NT;
    int k = kt * 32 + (lane >> 4) * 8 + (i & 7) + (i >> 3) * 16;
    int n = nt * 16 + (lane & 15);
    float v = (n < ncols) ? W[(size_t)l * wstride + (size_t)k * ncols + n] : 0.0f;
    __bf16 h = (__bf16)v;
    hi[idx] = h;
    lo[idx] = (__bf16)(v - (float)h);
}

// ---------------------------------------------------------------------------
// WMMA GEMM: Y[n x STRIDE] (+)= X[n x 128] @ Wpacked.  Block = 256 thr
// (8 waves), each wave owns a 16-row strip.  Three independent WMMA
// accumulator chains (hi*hi, lo*hi, hi*lo) merged at the end.
// ---------------------------------------------------------------------------
template <int NT, int STRIDE, bool ACCUM>
__global__ __launch_bounds__(256) void wmma_gemm_k(
    const float* __restrict__ X, const __bf16* __restrict__ Wh,
    const __bf16* __restrict__ Wl, float* __restrict__ Y, int n) {
    const int lane = threadIdx.x & 31;
    const int wid  = threadIdx.x >> 5;
    const int row0 = blockIdx.x * 128 + wid * 16;
    const int m  = lane & 15;
    const int kh = lane >> 4;

    int rc = row0 + m;
    if (rc > n - 1) rc = n - 1;             // clamp loads; stores are guarded
    const float* xr = X + (size_t)rc * FDIM;

    // Build A fragments (16-bit A 16x32 layout): lane m, half kh.
    v16bf ah[4], al[4];
#pragma unroll
    for (int kt = 0; kt < 4; ++kt) {
        int klo = kt * 32 + kh * 8;
        float4 G0 = *(const float4*)(xr + klo);
        float4 G1 = *(const float4*)(xr + klo + 4);
        float4 G2 = *(const float4*)(xr + klo + 16);
        float4 G3 = *(const float4*)(xr + klo + 20);
        float f[16] = {G0.x, G0.y, G0.z, G0.w, G1.x, G1.y, G1.z, G1.w,
                       G2.x, G2.y, G2.z, G2.w, G3.x, G3.y, G3.z, G3.w};
#pragma unroll
        for (int i = 0; i < 16; ++i) {
            __bf16 h = (__bf16)f[i];
            ah[kt][i] = h;
            al[kt][i] = (__bf16)(f[i] - (float)h);
        }
    }

#pragma unroll 1
    for (int nt = 0; nt < NT; ++nt) {
        v8f acc0 = {}, acc1 = {}, acc2 = {};
#pragma unroll
        for (int kt = 0; kt < 4; ++kt) {
            int fi = ((kt * NT + nt) * 32 + lane) * 16;
            v16bf bh = *(const v16bf*)(Wh + fi);
            v16bf bl = *(const v16bf*)(Wl + fi);
            acc0 = __builtin_amdgcn_wmma_f32_16x16x32_bf16(false, ah[kt], false, bh,
                                                           (short)0, acc0, false, false);
            acc1 = __builtin_amdgcn_wmma_f32_16x16x32_bf16(false, al[kt], false, bh,
                                                           (short)0, acc1, false, false);
            acc2 = __builtin_amdgcn_wmma_f32_16x16x32_bf16(false, ah[kt], false, bl,
                                                           (short)0, acc2, false, false);
        }
        v8f acc = acc0 + acc1 + acc2;
        // C/D layout: vgpr r -> row (r or r+8), col = lane&15.
        int col   = nt * 16 + (lane & 15);
        int rbase = row0 + (lane >> 4) * 8;
#pragma unroll
        for (int r = 0; r < 8; ++r) {
            int rr = rbase + r;
            if (rr < n) {
                size_t o = (size_t)rr * STRIDE + col;
                if (ACCUM) Y[o] += acc[r]; else Y[o] = acc[r];
            }
        }
    }
}

// ---------------------------------------------------------------------------
// agg = 2*dinv^2 * xw + bias   (self-loop term + conv bias)
// ---------------------------------------------------------------------------
__global__ void agg_init_k(const float* __restrict__ xw, const float* __restrict__ dinv,
                           const float* __restrict__ b, float* __restrict__ agg, int n) {
    size_t idx = (size_t)blockIdx.x * blockDim.x + threadIdx.x;
    if (idx >= (size_t)n * FDIM) return;
    int i = (int)(idx >> 7), f = (int)(idx & 127);
    float di = dinv[i];
    agg[idx] = 2.0f * di * di * xw[idx] + b[f];
}

// ---------------------------------------------------------------------------
// Edge scatter: one wave per edge. src/dst/coef wave-uniform (scalar loads);
// each lane moves one float4 of the 512B row; f32 atomics land in L2.
// ---------------------------------------------------------------------------
__global__ __launch_bounds__(256) void edge_agg_k(
    const int* __restrict__ src, const int* __restrict__ dst,
    const float* __restrict__ dinv, const float* __restrict__ xw,
    float* __restrict__ agg, int e) {
    const int lane = threadIdx.x & 31;
    int w = (int)((blockIdx.x * blockDim.x + threadIdx.x) >> 5);
    const int stride = (gridDim.x * blockDim.x) >> 5;
    for (; w < e; w += stride) {
        int s = src[w], d = dst[w];
        float c = dinv[s] * dinv[d];
        float4 v = *(const float4*)(xw + (size_t)s * FDIM + lane * 4);
        float* ap = agg + (size_t)d * FDIM + lane * 4;
        atomicAdd(ap + 0, c * v.x);
        atomicAdd(ap + 1, c * v.y);
        atomicAdd(ap + 2, c * v.z);
        atomicAdd(ap + 3, c * v.w);
        int wn = w + stride;
        if (wn < e)  // warm L2/L0 for next gather row
            __builtin_prefetch(xw + (size_t)src[wn] * FDIM + lane * 4, 0, 1);
    }
}

// ---------------------------------------------------------------------------
// BatchNorm (training stats) + ReLU
// ---------------------------------------------------------------------------
__global__ void bn_zero_k(float* sums) {
    int i = threadIdx.x;
    if (i < 256) sums[i] = 0.0f;
}

__global__ __launch_bounds__(128) void bn_stats_k(const float* __restrict__ a,
                                                  float* __restrict__ sums, int n) {
    int f  = threadIdx.x;            // channel
    int r0 = blockIdx.x * 512;
    int r1 = r0 + 512; if (r1 > n) r1 = n;
    float s = 0.0f, ss = 0.0f;
    for (int r = r0; r < r1; ++r) {
        float v = a[(size_t)r * FDIM + f];
        s += v; ss += v * v;
    }
    atomicAdd(&sums[f], s);
    atomicAdd(&sums[128 + f], ss);
}

__global__ void bn_fin_k(const float* __restrict__ sums, const float* __restrict__ g,
                         const float* __restrict__ b, float* __restrict__ ssh, int n) {
    int f = threadIdx.x;
    if (f >= 128) return;
    float inv_n = 1.0f / (float)n;
    float mu  = sums[f] * inv_n;
    float var = sums[128 + f] * inv_n - mu * mu;
    float sc  = g[f] * rsqrtf(var + 1e-5f);
    ssh[f]       = sc;
    ssh[128 + f] = b[f] - mu * sc;
}

__global__ void bn_apply_k(float* __restrict__ h, const float* __restrict__ ssh, int n) {
    size_t idx = (size_t)blockIdx.x * blockDim.x + threadIdx.x;
    if (idx >= (size_t)n * FDIM) return;
    int f = (int)(idx & 127);
    float v = h[idx] * ssh[f] + ssh[128 + f];
    h[idx] = fmaxf(v, 0.0f);
}

// ---------------------------------------------------------------------------
// Padded head accumulator init (sum of all fc biases; cols >= 40 are zero)
// and final log_softmax (reads padded [N,48], writes packed [N,40]).
// ---------------------------------------------------------------------------
__global__ void out_init_k(const float* __restrict__ fcb, float* __restrict__ outp, int n) {
    size_t idx = (size_t)blockIdx.x * blockDim.x + threadIdx.x;
    if (idx >= (size_t)n * CPAD) return;
    int c = (int)(idx % CPAD);
    float s = 0.0f;
    if (c < CDIM) {
#pragma unroll
        for (int l = 0; l < NLAYERS + 1; ++l) s += fcb[l * CDIM + c];
    }
    outp[idx] = s;
}

__global__ void logsoftmax_k(const float* __restrict__ outp, float* __restrict__ out, int n) {
    int i = blockIdx.x * blockDim.x + threadIdx.x;
    if (i >= n) return;
    const float* o = outp + (size_t)i * CPAD;
    float v[CDIM];
    float m = -3.4e38f;
#pragma unroll
    for (int c = 0; c < CDIM; ++c) { v[c] = o[c]; m = fmaxf(m, v[c]); }
    float s = 0.0f;
#pragma unroll
    for (int c = 0; c < CDIM; ++c) s += __expf(v[c] - m);
    float lse = m + __logf(s);
    float* q = out + (size_t)i * CDIM;
#pragma unroll
    for (int c = 0; c < CDIM; ++c) q[c] = v[c] - lse;
}

// ---------------------------------------------------------------------------
// Launch
// ---------------------------------------------------------------------------
extern "C" void kernel_launch(void* const* d_in, const int* in_sizes, int n_in,
                              void* d_out, int out_size, void* d_ws, size_t ws_size,
                              hipStream_t stream) {
    const float* x      = (const float*)d_in[0];
    const int*   eidx   = (const int*)d_in[1];
    const float* conv_w = (const float*)d_in[2];
    const float* conv_b = (const float*)d_in[3];
    const float* bn_g   = (const float*)d_in[4];
    const float* bn_b   = (const float*)d_in[5];
    const float* fc_w   = (const float*)d_in[6];
    const float* fc_b   = (const float*)d_in[7];

    const int N = in_sizes[0] / FDIM;
    const int E = in_sizes[1] / 2;
    const int* src = eidx;
    const int* dst = eidx + E;
    float* out = (float*)d_out;

    // Carve workspace
    char*  base = (char*)d_ws;
    size_t off  = 0;
    auto carve = [&](size_t bytes) -> void* {
        void* p = base + off;
        off = (off + bytes + 255) & ~(size_t)255;
        return p;
    };
    float*  hbuf  = (float*)carve((size_t)N * FDIM * 4);   // agg / activations
    float*  xw    = (float*)carve((size_t)N * FDIM * 4);   // h @ W
    float*  outp  = (float*)carve((size_t)N * CPAD * 4);   // padded head accumulator
    float*  dinv  = (float*)carve((size_t)N * 4);
    float*  sums  = (float*)carve(256 * 4);
    float*  ssh   = (float*)carve(256 * 4);
    const int convFrag = 4 * 8 * 32 * 16;                  // per-layer packed elems
    const int fcFrag   = 4 * 3 * 32 * 16;
    __bf16* cWh = (__bf16*)carve((size_t)NLAYERS * convFrag * 2);
    __bf16* cWl = (__bf16*)carve((size_t)NLAYERS * convFrag * 2);
    __bf16* fWh = (__bf16*)carve((size_t)(NLAYERS + 1) * fcFrag * 2);
    __bf16* fWl = (__bf16*)carve((size_t)(NLAYERS + 1) * fcFrag * 2);

    const int TB = 256;
    dim3 blk(TB);

    // --- normalization coefficients (shared across layers) ---
    deg_init_k<<<(N + TB - 1) / TB, blk, 0, stream>>>(dinv, N);
    deg_acc_k<<<(E + TB - 1) / TB, blk, 0, stream>>>(dst, dinv, E);
    deg_inv_k<<<(N + TB - 1) / TB, blk, 0, stream>>>(dinv, N);

    // --- pack weights into WMMA fragment order (bf16 hi/lo) ---
    pack_w_k<<<(NLAYERS * convFrag + TB - 1) / TB, blk, 0, stream>>>(
        conv_w, cWh, cWl, FDIM, 8, NLAYERS, FDIM * FDIM);
    pack_w_k<<<((NLAYERS + 1) * fcFrag + TB - 1) / TB, blk, 0, stream>>>(
        fc_w, fWh, fWl, CDIM, 3, NLAYERS + 1, FDIM * CDIM);

    // --- padded head accumulator = sum of fc biases ---
    out_init_k<<<(int)(((size_t)N * CPAD + TB - 1) / TB), blk, 0, stream>>>(fc_b, outp, N);

    const int gemmGrid = (N + 127) / 128;
    const int elemGrid = (int)(((size_t)N * FDIM + TB - 1) / TB);

    // head 0: x @ fc_w[0]
    wmma_gemm_k<3, CPAD, true><<<gemmGrid, blk, 0, stream>>>(x, fWh, fWl, outp, N);

    for (int l = 0; l < NLAYERS; ++l) {
        const float* hin = (l == 0) ? x : hbuf;
        // xw = h @ W_l
        wmma_gemm_k<8, FDIM, false><<<gemmGrid, blk, 0, stream>>>(
            hin, cWh + (size_t)l * convFrag, cWl + (size_t)l * convFrag, xw, N);
        // agg (into hbuf): self-loop + bias, then edge scatter
        agg_init_k<<<elemGrid, blk, 0, stream>>>(xw, dinv, conv_b + l * FDIM, hbuf, N);
        edge_agg_k<<<8192, blk, 0, stream>>>(src, dst, dinv, xw, hbuf, E);
        // batchnorm stats + apply (+ReLU), in place on hbuf
        bn_zero_k<<<1, blk, 0, stream>>>(sums);
        bn_stats_k<<<(N + 511) / 512, dim3(128), 0, stream>>>(hbuf, sums, N);
        bn_fin_k<<<1, dim3(128), 0, stream>>>(sums, bn_g + l * FDIM, bn_b + l * FDIM, ssh, N);
        bn_apply_k<<<elemGrid, blk, 0, stream>>>(hbuf, ssh, N);
        // head l+1
        wmma_gemm_k<3, CPAD, true><<<gemmGrid, blk, 0, stream>>>(
            hbuf, fWh + (size_t)(l + 1) * fcFrag, fWl + (size_t)(l + 1) * fcFrag, outp, N);
    }

    logsoftmax_k<<<(N + TB - 1) / TB, blk, 0, stream>>>(outp, out, N);
}
